// MultiHeadAttention_50964081935044
// MI455X (gfx1250) — matrix operand order repaired and checked by reading, not compiled, not run on hardware
//
#include <hip/hip_runtime.h>
#include <hip/hip_bf16.h>

// ---------------------------------------------------------------------------
// Types for CDNA5 WMMA (wave32, 16x16x32 bf16 -> f32)
// ---------------------------------------------------------------------------
typedef __bf16 bf16;
typedef __attribute__((ext_vector_type(8)))  bf16  v8bf;
typedef __attribute__((ext_vector_type(16))) bf16  v16bf;
typedef __attribute__((ext_vector_type(8)))  float v8f;

union FragU { v16bf v; v8bf h[2]; };

// A-matrix fragment (16x32, 16-bit): lane holds row (lane&15);
// K elements: base=(lane>=16?8:0): e0..7 -> K=base..base+7, e8..15 -> K=base+16..base+23
__device__ __forceinline__ v16bf load_frag_a(const bf16* p, int ld, int lane) {
    const int row  = lane & 15;
    const int base = (lane & 16) ? 8 : 0;
    const bf16* q = p + (size_t)row * ld + base;
    FragU f;
    f.h[0] = *(const v8bf*)(q);
    f.h[1] = *(const v8bf*)(q + 16);
    return f.v;
}

// B-matrix fragment (32x16, 16-bit): lane holds column N=(lane&15);
// K elements: base=(lane>=16?8:0): e0..7 -> K=base..base+7, e8..15 -> K=base+16..base+23
// Source stored K-contiguous per column (i.e. row-major [N][K]).
__device__ __forceinline__ v16bf load_frag_b(const bf16* p, int ld, int lane) {
    const int col  = lane & 15;
    const int base = (lane & 16) ? 8 : 0;
    const bf16* q = p + (size_t)col * ld + base;
    FragU f;
    f.h[0] = *(const v8bf*)(q);
    f.h[1] = *(const v8bf*)(q + 16);
    return f.v;
}

__device__ __forceinline__ v8f wmma_bf16(v16bf a, v16bf b, v8f c) {
    return __builtin_amdgcn_wmma_f32_16x16x32_bf16(
        /*neg_a=*/false, a, /*neg_b=*/false, b,
        /*c_mod=*/(short)0, c, /*reuse_a=*/false, /*reuse_b=*/false);
}

// ---------------------------------------------------------------------------
// CDNA5 async global->LDS copy (ASYNCcnt-tracked), with safe fallback
// ---------------------------------------------------------------------------
#if __has_builtin(__builtin_amdgcn_global_load_async_to_lds_b128)
#define HAVE_ASYNC_LDS 1
#else
#define HAVE_ASYNC_LDS 0
#endif

#if __has_builtin(__builtin_amdgcn_s_wait_asynccnt)
#define WAIT_ASYNC() __builtin_amdgcn_s_wait_asynccnt(0)
#else
#define WAIT_ASYNC() asm volatile("s_wait_asynccnt 0" ::: "memory")
#endif

typedef int v4i_vs __attribute__((vector_size(16)));   // matches builtin param type
typedef __attribute__((address_space(1))) v4i_vs* gptr_b128;
typedef __attribute__((address_space(3))) v4i_vs* lptr_b128;

__device__ __forceinline__ void async_cp16(const bf16* g, bf16* l) {
#if HAVE_ASYNC_LDS
    __builtin_amdgcn_global_load_async_to_lds_b128((gptr_b128)g, (lptr_b128)l, 0, 0);
#else
    *(uint4*)l = *(const uint4*)g;   // vmem load + ds store fallback
#endif
}

// ---------------------------------------------------------------------------
// Problem constants
// ---------------------------------------------------------------------------
#define BSZ    2
#define SEQ    2048
#define DMODEL 1024
#define NHEADS 16
#define DK     64
#define MROWS  (BSZ * SEQ)   // 4096

// ---------------------------------------------------------------------------
// f32 -> bf16 conversion
// ---------------------------------------------------------------------------
__global__ __launch_bounds__(256) void cvt_f32_bf16(const float* __restrict__ in,
                                                    bf16* __restrict__ out, int n) {
    int i = blockIdx.x * 256 + threadIdx.x;
    if (i < n) out[i] = (bf16)in[i];
}

// ---------------------------------------------------------------------------
// WMMA GEMM: C[M,N] = A[M,K] * W[N,K]^T   (torch Linear)
// Each wave computes a 32(M) x 64(N) tile (2 A-frags x 4 B-frags -> 8 WMMA/step).
// Block = 8 waves -> 256(M) x 64(N).
// mode 0: write f32 row-major [M,N]
// mode 1: write bf16 [b,h,s,d]   (Q,K head layout)
// mode 2: write bf16 [b,h,d,s]   (V transposed head layout)
// ---------------------------------------------------------------------------
__global__ __launch_bounds__(256) void gemm_wmma_bf16(
    const bf16* __restrict__ A, const bf16* __restrict__ W,
    float* __restrict__ outF, bf16* __restrict__ outB,
    int M, int N, int K, int mode)
{
    const int lane = threadIdx.x & 31;
    const int wave = threadIdx.x >> 5;
    const int m0 = blockIdx.x * 256 + wave * 32;
    const int n0 = blockIdx.y * 64;

    v8f acc[2][4] = {};
    for (int k0 = 0; k0 < K; k0 += 32) {
        v16bf a0 = load_frag_a(A + (size_t)m0 * K + k0,        K, lane);
        v16bf a1 = load_frag_a(A + (size_t)(m0 + 16) * K + k0, K, lane);
        #pragma unroll
        for (int t = 0; t < 4; ++t) {
            v16bf b = load_frag_b(W + (size_t)(n0 + 16 * t) * K + k0, K, lane);
            acc[0][t] = wmma_bf16(a0, b, acc[0][t]);
            acc[1][t] = wmma_bf16(a1, b, acc[1][t]);
        }
    }

    const int ncol  = lane & 15;
    const int mbase = (lane & 16) ? 8 : 0;
    #pragma unroll
    for (int u = 0; u < 2; ++u) {
        #pragma unroll
        for (int t = 0; t < 4; ++t) {
            #pragma unroll
            for (int r = 0; r < 8; ++r) {
                const int m = m0 + 16 * u + mbase + r;
                const int n = n0 + 16 * t + ncol;
                const float v = acc[u][t][r];
                if (mode == 0) {
                    outF[(size_t)m * N + n] = v;
                } else {
                    const int b = m / SEQ, s = m % SEQ;
                    const int h = n >> 6,  d = n & 63;
                    const size_t idx = (mode == 1)
                        ? ((((size_t)b * NHEADS + h) * SEQ + s) * DK + d)
                        : ((((size_t)b * NHEADS + h) * DK + d) * SEQ + s);
                    outB[idx] = (bf16)v;
                }
            }
        }
    }
}

// ---------------------------------------------------------------------------
// Cooperative async staging of one 32-key tile:
//   K tile: 32 rows x 64 d  (row-major, ld=64)  = 4 KB
//   V tile: 64 rows(d) x 32 keys (row-major, ld=32) = 4 KB
// 128 threads, each moving 4 x 16B chunks.
// ---------------------------------------------------------------------------
__device__ __forceinline__ void stage_tiles(const bf16* Kp, const bf16* Vp, int j,
                                            bf16* ldsK, bf16* ldsV, int tid) {
    #pragma unroll
    for (int i = 0; i < 2; ++i) {               // K tile: 256 chunks
        const int c = tid + i * 128;
        const int row = c >> 3, col = (c & 7) * 8;
        async_cp16(Kp + (size_t)(j + row) * DK + col, ldsK + c * 8);
    }
    #pragma unroll
    for (int i = 0; i < 2; ++i) {               // V tile: 256 chunks
        const int c = tid + i * 128;
        const int d = c >> 2, col = (c & 3) * 8;
        async_cp16(Vp + (size_t)d * SEQ + j + col, ldsV + c * 8);
    }
}

// ---------------------------------------------------------------------------
// Flash attention (non-causal, full softmax), bf16 WMMA, f32 accumulation.
// Block = 4 waves = 64 queries of one (b,h); each wave owns 16 queries.
// K/V tiles are block-shared, double-buffered in LDS via async global->LDS.
// Qh,Kh: [b,h,s,d] bf16;  VhT: [b,h,d,s] bf16;  ctx: [b,s,h*64+d] bf16.
// ---------------------------------------------------------------------------
__global__ __launch_bounds__(128) void flash_attn_wmma(
    const bf16* __restrict__ Qh, const bf16* __restrict__ Kh,
    const bf16* __restrict__ VhT, bf16* __restrict__ ctx)
{
    __shared__ bf16 ldsK[2][32 * DK];   // 2 x 4 KB, [key][d]
    __shared__ bf16 ldsV[2][DK * 32];   // 2 x 4 KB, [d][key]
    __shared__ bf16 ldsP[4][16][32];    // per-wave P tile, 4 KB

    const int tid  = threadIdx.x;
    const int lane = tid & 31;
    const int wave = tid >> 5;
    const int nqblk = SEQ / 64;
    const int qblk = blockIdx.x % nqblk;
    const int bh   = blockIdx.x / nqblk;
    const int b = bh / NHEADS, h = bh % NHEADS;
    const int q0 = qblk * 64 + wave * 16;

    const bf16* Qp = Qh  + ((((size_t)b * NHEADS + h) * SEQ) + q0) * DK;
    const bf16* Kp = Kh  + (((size_t)b * NHEADS + h) * SEQ) * DK;
    const bf16* Vp = VhT + (((size_t)b * NHEADS + h) * DK) * (size_t)SEQ;

    // Q fragments (d = 0..31, 32..63), loaded once from global
    const v16bf qa0 = load_frag_a(Qp + 0,  DK, lane);
    const v16bf qa1 = load_frag_a(Qp + 32, DK, lane);

    v8f acc[4] = {};
    float mrow[8], lrow[8];
    #pragma unroll
    for (int r = 0; r < 8; ++r) { mrow[r] = -1e30f; lrow[r] = 0.f; }

    const float scale = 0.125f;   // 1/sqrt(64)
    const int mbase = (lane & 16) ? 8 : 0;
    const int col0 = lane & 15;

    // prologue: stage tile j=0
    stage_tiles(Kp, Vp, 0, ldsK[0], ldsV[0], tid);
#if HAVE_ASYNC_LDS
    WAIT_ASYNC();
#endif
    __syncthreads();

    int cur = 0;
    for (int j = 0; j < SEQ; j += 32) {
        const bf16* Kt = ldsK[cur];
        const bf16* Vt = ldsV[cur];

        // prefetch next tile into the other buffer (overlaps with compute)
        if (j + 32 < SEQ)
            stage_tiles(Kp, Vp, j + 32, ldsK[cur ^ 1], ldsV[cur ^ 1], tid);

        // --- scores S = (Q K^T) * scale : two 16x16 tiles (keys j..j+15, j+16..j+31)
        v8f s0 = {}, s1 = {};
        {
            v16bf bk;
            bk = load_frag_b(Kt + 0 * DK + 0,   DK, lane); s0 = wmma_bf16(qa0, bk, s0);
            bk = load_frag_b(Kt + 0 * DK + 32,  DK, lane); s0 = wmma_bf16(qa1, bk, s0);
            bk = load_frag_b(Kt + 16 * DK + 0,  DK, lane); s1 = wmma_bf16(qa0, bk, s1);
            bk = load_frag_b(Kt + 16 * DK + 32, DK, lane); s1 = wmma_bf16(qa1, bk, s1);
        }

        // --- streaming softmax update (row stats via 16-lane shuffles)
        float alpha[8];
        #pragma unroll
        for (int r = 0; r < 8; ++r) {
            const float a = s0[r] * scale;
            const float c = s1[r] * scale;
            float mx = fmaxf(a, c);
            mx = fmaxf(mx, __shfl_xor(mx, 1, 16));
            mx = fmaxf(mx, __shfl_xor(mx, 2, 16));
            mx = fmaxf(mx, __shfl_xor(mx, 4, 16));
            mx = fmaxf(mx, __shfl_xor(mx, 8, 16));
            const float mnew = fmaxf(mrow[r], mx);
            alpha[r] = __expf(mrow[r] - mnew);
            const float p0 = __expf(a - mnew);
            const float p1 = __expf(c - mnew);
            float rs = p0 + p1;
            rs += __shfl_xor(rs, 1, 16);
            rs += __shfl_xor(rs, 2, 16);
            rs += __shfl_xor(rs, 4, 16);
            rs += __shfl_xor(rs, 8, 16);
            lrow[r] = lrow[r] * alpha[r] + rs;
            mrow[r] = mnew;
            // C-layout -> LDS row-major P (for A-fragment reload)
            ldsP[wave][mbase + r][col0]      = (bf16)p0;
            ldsP[wave][mbase + r][16 + col0] = (bf16)p1;
        }

        #pragma unroll
        for (int t = 0; t < 4; ++t)
            #pragma unroll
            for (int r = 0; r < 8; ++r)
                acc[t][r] *= alpha[r];

        // same-wave LDS ops are in-order; explicit wait for safety
        asm volatile("s_wait_dscnt 0" ::: "memory");

        // --- ctx += P(16x32) @ V(32x64)
        const v16bf pa = load_frag_a(&ldsP[wave][0][0], 32, lane);
        #pragma unroll
        for (int t = 0; t < 4; ++t) {
            v16bf bv = load_frag_b(Vt + (size_t)(16 * t) * 32, 32, lane);
            acc[t] = wmma_bf16(pa, bv, acc[t]);
        }

        // prefetch complete + all waves done reading 'cur' before flipping
#if HAVE_ASYNC_LDS
        WAIT_ASYNC();
#endif
        __syncthreads();
        cur ^= 1;
    }

    // --- normalize and write ctx [b, s, h*64 + d]
    #pragma unroll
    for (int r = 0; r < 8; ++r) {
        const float inv = 1.0f / lrow[r];
        const int q = q0 + mbase + r;
        #pragma unroll
        for (int t = 0; t < 4; ++t) {
            const int d = 16 * t + col0;
            ctx[((size_t)b * SEQ + q) * DMODEL + h * DK + d] = (bf16)(acc[t][r] * inv);
        }
    }
}

// ---------------------------------------------------------------------------
// Residual + LayerNorm: out = LN(query + attn_out) * gamma + beta
// One block per row of DMODEL.
// ---------------------------------------------------------------------------
__global__ __launch_bounds__(256) void ln_residual(
    const float* __restrict__ x, const float* __restrict__ y,
    const float* __restrict__ gamma, const float* __restrict__ beta,
    float* __restrict__ out)
{
    __shared__ float red[256];
    const int row = blockIdx.x;
    const float* xr = x + (size_t)row * DMODEL;
    const float* yr = y + (size_t)row * DMODEL;

    float s = 0.f, s2 = 0.f;
    for (int i = threadIdx.x; i < DMODEL; i += 256) {
        const float v = xr[i] + yr[i];
        s += v; s2 += v * v;
    }
    red[threadIdx.x] = s; __syncthreads();
    for (int off = 128; off > 0; off >>= 1) {
        if (threadIdx.x < off) red[threadIdx.x] += red[threadIdx.x + off];
        __syncthreads();
    }
    const float mean = red[0] / DMODEL;
    __syncthreads();
    red[threadIdx.x] = s2; __syncthreads();
    for (int off = 128; off > 0; off >>= 1) {
        if (threadIdx.x < off) red[threadIdx.x] += red[threadIdx.x + off];
        __syncthreads();
    }
    const float var  = red[0] / DMODEL - mean * mean;
    const float rstd = rsqrtf(var + 1e-5f);
    for (int i = threadIdx.x; i < DMODEL; i += 256) {
        const float v = xr[i] + yr[i];
        out[(size_t)row * DMODEL + i] = (v - mean) * rstd * gamma[i] + beta[i];
    }
}

// ---------------------------------------------------------------------------
// Host-side orchestration
// ---------------------------------------------------------------------------
extern "C" void kernel_launch(void* const* d_in, const int* in_sizes, int n_in,
                              void* d_out, int out_size, void* d_ws, size_t ws_size,
                              hipStream_t stream) {
    const float* q_in  = (const float*)d_in[0];
    const float* k_in  = (const float*)d_in[1];
    const float* v_in  = (const float*)d_in[2];
    const float* wq    = (const float*)d_in[3];
    const float* wk    = (const float*)d_in[4];
    const float* wv    = (const float*)d_in[5];
    const float* wo    = (const float*)d_in[6];
    const float* gamma = (const float*)d_in[7];
    const float* beta  = (const float*)d_in[8];
    float* out = (float*)d_out;

    const size_t MB = 1024u * 1024u;
    char* ws = (char*)d_ws;
    // Workspace layout (56 MB total):
    bf16* xq  = (bf16*)(ws + 0 * MB);    // 8 MB  (reused as ctx after QKV proj)
    bf16* xk  = (bf16*)(ws + 8 * MB);    // 8 MB  (xk+xv reused as f32 attn_out)
    bf16* xv  = (bf16*)(ws + 16 * MB);   // 8 MB
    bf16* wqb = (bf16*)(ws + 24 * MB);   // 2 MB
    bf16* wkb = (bf16*)(ws + 26 * MB);   // 2 MB
    bf16* wvb = (bf16*)(ws + 28 * MB);   // 2 MB
    bf16* wob = (bf16*)(ws + 30 * MB);   // 2 MB
    bf16* Qh  = (bf16*)(ws + 32 * MB);   // 8 MB [b,h,s,d]
    bf16* Kh  = (bf16*)(ws + 40 * MB);   // 8 MB [b,h,s,d]
    bf16* VhT = (bf16*)(ws + 48 * MB);   // 8 MB [b,h,d,s]
    bf16*  ctx      = xq;                // 8 MB bf16 [B*S, DMODEL]
    float* attn_out = (float*)xk;        // 16 MB f32 [B*S, DMODEL]

    const int nAct = MROWS * DMODEL;     // 4 Mi elements
    const int nW   = DMODEL * DMODEL;    // 1 Mi elements

    // 1) convert activations + weights to bf16
    cvt_f32_bf16<<<(nAct + 255) / 256, 256, 0, stream>>>(q_in, xq, nAct);
    cvt_f32_bf16<<<(nAct + 255) / 256, 256, 0, stream>>>(k_in, xk, nAct);
    cvt_f32_bf16<<<(nAct + 255) / 256, 256, 0, stream>>>(v_in, xv, nAct);
    cvt_f32_bf16<<<(nW + 255) / 256, 256, 0, stream>>>(wq, wqb, nW);
    cvt_f32_bf16<<<(nW + 255) / 256, 256, 0, stream>>>(wk, wkb, nW);
    cvt_f32_bf16<<<(nW + 255) / 256, 256, 0, stream>>>(wv, wvb, nW);
    cvt_f32_bf16<<<(nW + 255) / 256, 256, 0, stream>>>(wo, wob, nW);

    // 2) Q/K/V projections (WMMA GEMM), head-friendly output layouts
    dim3 gg(MROWS / 256, DMODEL / 64);
    gemm_wmma_bf16<<<gg, 256, 0, stream>>>(xq, wqb, nullptr, Qh,  MROWS, DMODEL, DMODEL, 1);
    gemm_wmma_bf16<<<gg, 256, 0, stream>>>(xk, wkb, nullptr, Kh,  MROWS, DMODEL, DMODEL, 1);
    gemm_wmma_bf16<<<gg, 256, 0, stream>>>(xv, wvb, nullptr, VhT, MROWS, DMODEL, DMODEL, 2);

    // 3) flash attention: grid = B*H*(S/64), 4 waves/block
    flash_attn_wmma<<<BSZ * NHEADS * (SEQ / 64), 128, 0, stream>>>(Qh, Kh, VhT, ctx);

    // 4) output projection (f32 out)
    gemm_wmma_bf16<<<gg, 256, 0, stream>>>(ctx, wob, attn_out, nullptr, MROWS, DMODEL, DMODEL, 0);

    // 5) residual + LayerNorm
    ln_residual<<<MROWS, 256, 0, stream>>>(q_in, attn_out, gamma, beta, out);
}
